// MultiHeadAttention_16535624090489
// MI455X (gfx1250) — matrix operand order repaired and checked by reading, not compiled, not run on hardware
//
#include <hip/hip_runtime.h>
#include <hip/hip_bf16.h>

typedef __attribute__((ext_vector_type(16))) _Float16 v16h;
typedef __attribute__((ext_vector_type(8)))  _Float16 v8h;
typedef __attribute__((ext_vector_type(4)))  _Float16 v4h;
typedef __attribute__((ext_vector_type(8)))  float    v8f;
typedef __attribute__((ext_vector_type(4)))  unsigned int v4u;
typedef __attribute__((ext_vector_type(8)))  int      v8i;
typedef __attribute__((ext_vector_type(4)))  int      v4i;

#if __has_builtin(__builtin_amdgcn_tensor_load_to_lds) && \
    __has_builtin(__builtin_amdgcn_s_wait_tensorcnt)
#define HAS_TDM 1
#else
#define HAS_TDM 0
#endif

static __device__ __forceinline__ v16h cat8(v8h lo, v8h hi) {
    return __builtin_shufflevector(lo, hi, 0,1,2,3,4,5,6,7,8,9,10,11,12,13,14,15);
}

// DPP16 16-lane butterflies (stay within each 16-lane half of the wave).
#define DPP_F(x, ctrl) \
    __int_as_float(__builtin_amdgcn_update_dpp(0, __float_as_int(x), (ctrl), 0xf, 0xf, true))

static __device__ __forceinline__ float redmax16(float t) {
    t = fmaxf(t, DPP_F(t, 0x0B1));   // quad_perm xor1
    t = fmaxf(t, DPP_F(t, 0x04E));   // quad_perm xor2
    t = fmaxf(t, DPP_F(t, 0x141));   // row_half_mirror
    t = fmaxf(t, DPP_F(t, 0x140));   // row_mirror
    return t;
}
static __device__ __forceinline__ float redsum16(float t) {
    t += DPP_F(t, 0x0B1);
    t += DPP_F(t, 0x04E);
    t += DPP_F(t, 0x141);
    t += DPP_F(t, 0x140);
    return t;
}

#if HAS_TDM
// TDM: DMA a 64x64 f16 tile (row stride 1024 elements) from global to LDS.
// D# per CDNA5 ISA 8.3/8.4: group0 = {count=1 | type=2 | lds | gaddr},
// group1 = {data_size=2B, tensor_dim0/1=64, tile 64x64, dim0_stride=1024}.
// Groups 2/3 zero (2D tensor). clang-23 6-arg builtin.
static __device__ __forceinline__ void tdm_load_tile_64x64(
    const _Float16* gsrc, unsigned int lds_off)
{
    const unsigned long long ga = (unsigned long long)(uintptr_t)gsrc;
    v4u g0 = { 1u,                                     // count=1, user mode
               lds_off,                                // lds_addr
               (unsigned int)ga,                       // global_addr[31:0]
               (unsigned int)((ga >> 32) & 0x1FFFFFFu) // global_addr[56:32]
                   | (2u << 30) };                     // type=2 ("image")
    v8i g1 = { 0x00010000,       // wg_mask=0, data_size=1 (2 bytes)
               (64 << 16),       // tensor_dim0 = 64 (low 16 at bits 63:48)
               (64 << 16),       // tensor_dim0 hi=0, tensor_dim1 = 64 (low 16)
               (64 << 16),       // tensor_dim1 hi=0, tile_dim0 = 64
               64,               // tile_dim1 = 64, tile_dim2 = 0
               1024,             // tensor_dim0_stride = 1024 elements
               0, 0 };           // dim0_stride hi, dim1_stride = 0
    v4i z4 = { 0, 0, 0, 0 };
    v8i z8 = { 0, 0, 0, 0, 0, 0, 0, 0 };
    __builtin_amdgcn_tensor_load_to_lds(g0, g1, z4, z4, z8, 0);
}
#endif

// ---------------------------------------------------------------------------
// GEMM: out[m,n] = sum_k X[m,k] * W[n,k] + bias[n]   (nn.Linear, W row-major [N,K])
// M=8192, N=K=1024 fixed. Block: 256 thr (8 waves), tile 128x128,
// wave tile 64x32 (4x2 WMMA tiles), K-step 32.
// ---------------------------------------------------------------------------
template<typename TX, bool OUT_F16>
__global__ __launch_bounds__(256) void gemm_bias_kernel(
    const TX* __restrict__ X, const float* __restrict__ W,
    const float* __restrict__ bias,
    _Float16* __restrict__ outH, float* __restrict__ outF)
{
    constexpr int K = 1024, N = 1024;
    constexpr int BM = 128, BN = 128, BK = 32, LDT = 40; // 40 halves = 80B stride
    __shared__ _Float16 sA[BM * LDT];
    __shared__ _Float16 sB[BN * LDT];

    const int tid  = threadIdx.x;
    const int lane = tid & 31;
    const int wave = tid >> 5;
    const int l15  = lane & 15;
    const int lh   = lane >> 4;
    const int wm   = (wave & 1) * 64;    // 2 waves along M, 64 rows each
    const int wn   = (wave >> 1) * 32;   // 4 waves along N, 32 cols each
    const int m0   = blockIdx.x * BM;
    const int n0   = blockIdx.y * BN;

    v8f acc[4][2] = {};

    for (int k0 = 0; k0 < K; k0 += BK) {
        {   // stage A tile (128x32) -> f16 LDS, vectorized 16B stores
            const int row = tid >> 1;
            const int cg  = (tid & 1) * 16;
            const TX* src = X + (size_t)(m0 + row) * K + k0 + cg;
            v8h t0, t1;
            #pragma unroll
            for (int i = 0; i < 8; ++i) { t0[i] = (_Float16)src[i]; t1[i] = (_Float16)src[8 + i]; }
            _Float16* dst = &sA[row * LDT + cg];
            *(v8h*)(dst)     = t0;
            *(v8h*)(dst + 8) = t1;
        }
        {   // stage B tile (128x32) from W[n][k] -> f16 LDS
            const int row = tid >> 1;
            const int cg  = (tid & 1) * 16;
            const float* src = W + (size_t)(n0 + row) * K + k0 + cg;
            v8h t0, t1;
            #pragma unroll
            for (int i = 0; i < 8; ++i) { t0[i] = (_Float16)src[i]; t1[i] = (_Float16)src[8 + i]; }
            _Float16* dst = &sB[row * LDT + cg];
            *(v8h*)(dst)     = t0;
            *(v8h*)(dst + 8) = t1;
        }
        __syncthreads();

        v16h a[4];
        #pragma unroll
        for (int i = 0; i < 4; ++i) {
            const _Float16* base = &sA[(wm + i * 16 + l15) * LDT];
            v8h lo = *(const v8h*)(base + lh * 8);
            v8h hi = *(const v8h*)(base + 16 + lh * 8);
            a[i] = cat8(lo, hi);
        }
        v16h b[2];
        #pragma unroll
        for (int j = 0; j < 2; ++j) {
            const _Float16* base = &sB[(wn + j * 16 + l15) * LDT];
            v8h lo = *(const v8h*)(base + lh * 8);
            v8h hi = *(const v8h*)(base + 16 + lh * 8);
            b[j] = cat8(lo, hi);
        }
        #pragma unroll
        for (int i = 0; i < 4; ++i)
            #pragma unroll
            for (int j = 0; j < 2; ++j)
                acc[i][j] = __builtin_amdgcn_wmma_f32_16x16x32_f16(
                    false, a[i], false, b[j], (short)0, acc[i][j], false, false);
        __syncthreads();
    }

    #pragma unroll
    for (int j = 0; j < 2; ++j) {
        const int n  = n0 + wn + j * 16 + l15;
        const float bv = bias[n];
        #pragma unroll
        for (int i = 0; i < 4; ++i) {
            #pragma unroll
            for (int r = 0; r < 8; ++r) {
                const int m = m0 + wm + i * 16 + r + 8 * lh;
                const float v = acc[i][j][r] + bv;
                if constexpr (OUT_F16) outH[(size_t)m * N + n] = (_Float16)v;
                else                   outF[(size_t)m * N + n] = v;
            }
        }
    }
}

// ---------------------------------------------------------------------------
// Flash attention over Qp/Kp/Vp in [B,S,D] f16 (head h at column h*64).
// Block = 8 waves = 128 query rows of one (b,h). Key blocks of 64.
// K tiles double-buffered via the Tensor Data Mover: the DMA for tile kb+1 is
// issued before processing tile kb, so the 8 KB transfer overlaps a full
// iteration of WMMA+softmax. s_wait_tensorcnt 1 ensures the current tile has
// landed (TDM ops from one wave complete in order). Softmax in log2 domain.
// ---------------------------------------------------------------------------
__global__ __launch_bounds__(256) void attn_kernel(
    const _Float16* __restrict__ Qp, const _Float16* __restrict__ Kp,
    const _Float16* __restrict__ Vp, _Float16* __restrict__ Op)
{
    constexpr int S = 2048, D = 1024, HD = 64;
    constexpr int KB = 64;                       // keys per block iteration
    constexpr int NB = S / KB;                   // 32 key blocks
    constexpr int LDP = 72;                      // P stage stride (144B, 16B-aligned)
#if HAS_TDM
    __shared__ _Float16 sK[2][KB * HD];          // K tile ping-pong  (16 KB)
#endif
    __shared__ _Float16 sVT[HD * KB];            // V^T  [dim][key]    (8 KB)
    __shared__ _Float16 sP[8 * 16 * LDP];        // per-wave P stage  (18 KB)

    const int tid  = threadIdx.x;
    const int lane = tid & 31;
    const int wave = tid >> 5;
    const int l15  = lane & 15;
    const int lh   = lane >> 4;

    const int bh = blockIdx.y;                   // b*16 + h
    const int b  = bh >> 4, h = bh & 15;
    const size_t headoff = (size_t)b * S * D + (size_t)h * HD;
    const int q0 = blockIdx.x * 128 + wave * 16;

    // Q fragments (16x64 -> two K=32 frags), pre-scaled by 0.125*log2(e)
    v16h qa[2];
    {
        const _Float16* qrow = Qp + headoff + (size_t)(q0 + l15) * D;
        #pragma unroll
        for (int c = 0; c < 2; ++c) {
            v8h lo = *(const v8h*)(qrow + c * 32 + lh * 8);
            v8h hi = *(const v8h*)(qrow + c * 32 + 16 + lh * 8);
            v16h t = cat8(lo, hi);
            t *= (_Float16)(0.125f * 1.44269504f);
            qa[c] = t;
        }
    }

    v8f o[4] = {};
    v8f mrow, lrow;
    #pragma unroll
    for (int i = 0; i < 8; ++i) { mrow[i] = -3.0e38f; lrow[i] = 0.0f; }

    _Float16* myP = &sP[wave * 16 * LDP];
#if HAS_TDM
    const unsigned int sK_lds0 = (unsigned int)(uintptr_t)&sK[0][0];
    const unsigned int sK_lds1 = (unsigned int)(uintptr_t)&sK[1][0];
    if (wave == 0)   // prologue: DMA tile 0 into buffer 0
        tdm_load_tile_64x64(Kp + headoff, sK_lds0);
#endif

    for (int kb = 0; kb < NB; ++kb) {
        const int cur = kb & 1;
#if HAS_TDM
        if (wave == 0 && kb + 1 < NB)   // issue next tile into the other buffer
            tdm_load_tile_64x64(Kp + headoff + (size_t)((kb + 1) * KB) * D,
                                cur ? sK_lds0 : sK_lds1);
#endif
        // stage V^T: 4 keys x 4 dims per thread -> 4 coalesced b64 loads + 4 b64 LDS stores
        {
            const int kg  = tid >> 4;            // 0..15 -> keys 4*kg..4*kg+3
            const int dgi = tid & 15;            // dims dgi*4..dgi*4+3
            const _Float16* vb = Vp + headoff + (size_t)(kb * KB + 4 * kg) * D + dgi * 4;
            v4h r0 = *(const v4h*)(vb);
            v4h r1 = *(const v4h*)(vb + D);
            v4h r2 = *(const v4h*)(vb + 2 * D);
            v4h r3 = *(const v4h*)(vb + 3 * D);
            #pragma unroll
            for (int d = 0; d < 4; ++d) {
                v4h t = { r0[d], r1[d], r2[d], r3[d] };
                *(v4h*)&sVT[(dgi * 4 + d) * KB + 4 * kg] = t;
            }
        }
#if HAS_TDM
        if (wave == 0) {   // current tile landed; next may remain in flight
            if (kb + 1 < NB) __builtin_amdgcn_s_wait_tensorcnt(1);
            else             __builtin_amdgcn_s_wait_tensorcnt(0);
        }
#endif
        __syncthreads();

        // scores: S[16,64] = Qscaled[16,64] x K^T[64,64] -> four 16x16 D tiles
        v8f s[4];
        #pragma unroll
        for (int nt = 0; nt < 4; ++nt) {
#if HAS_TDM
            const _Float16* krow = &sK[cur][(nt * 16 + l15) * HD];
#else
            const _Float16* krow =
                Kp + headoff + (size_t)(kb * KB + nt * 16 + l15) * D;
#endif
            v8f c = {};
            #pragma unroll
            for (int cc = 0; cc < 2; ++cc) {
                v8h lo = *(const v8h*)(krow + cc * 32 + lh * 8);
                v8h hi = *(const v8h*)(krow + cc * 32 + 16 + lh * 8);
                c = __builtin_amdgcn_wmma_f32_16x16x32_f16(
                    false, qa[cc], false, cat8(lo, hi), (short)0, c, false, false);
            }
            s[nt] = c;
        }

        // online softmax in log2 domain (row of element i is i + 8*lh)
        v8f mnew = mrow, alpha;
        #pragma unroll
        for (int i = 0; i < 8; ++i) {
            float t = fmaxf(fmaxf(s[0][i], s[1][i]), fmaxf(s[2][i], s[3][i]));
            mnew[i] = fmaxf(mnew[i], redmax16(t));
        }
        v8f p[4];
        #pragma unroll
        for (int i = 0; i < 8; ++i) {
            alpha[i] = exp2f(mrow[i] - mnew[i]);
            #pragma unroll
            for (int nt = 0; nt < 4; ++nt) p[nt][i] = exp2f(s[nt][i] - mnew[i]);
        }
        mrow = mnew;
        #pragma unroll
        for (int i = 0; i < 8; ++i) {
            float rs = redsum16((p[0][i] + p[1][i]) + (p[2][i] + p[3][i]));
            lrow[i] = lrow[i] * alpha[i] + rs;
        }
        #pragma unroll
        for (int t = 0; t < 4; ++t) o[t] *= alpha;

        // stage P (C/D layout -> LDS [row][col]) then reload in A layout
        #pragma unroll
        for (int nt = 0; nt < 4; ++nt)
            #pragma unroll
            for (int i = 0; i < 8; ++i)
                myP[(i + 8 * lh) * LDP + nt * 16 + l15] = (_Float16)p[nt][i];
        __builtin_amdgcn_wave_barrier();

        v16h pa[2];
        #pragma unroll
        for (int cc = 0; cc < 2; ++cc) {
            const _Float16* base = &myP[l15 * LDP + cc * 32];
            v8h lo = *(const v8h*)(base + lh * 8);
            v8h hi = *(const v8h*)(base + 16 + lh * 8);
            pa[cc] = cat8(lo, hi);
        }

        // O[16,64] += P[16,64] x V[64,64]
        #pragma unroll
        for (int ct = 0; ct < 4; ++ct) {
            #pragma unroll
            for (int cc = 0; cc < 2; ++cc) {
                const _Float16* base = &sVT[(ct * 16 + l15) * KB + cc * 32];
                v8h lo = *(const v8h*)(base + lh * 8);
                v8h hi = *(const v8h*)(base + 16 + lh * 8);
                o[ct] = __builtin_amdgcn_wmma_f32_16x16x32_f16(
                    false, pa[cc], false, cat8(lo, hi), (short)0, o[ct], false, false);
            }
        }
        __syncthreads();   // all reads of sK[cur]/sVT done before restage
    }

    // normalize and store merged-head f16 output
    v8f inv;
    #pragma unroll
    for (int i = 0; i < 8; ++i) inv[i] = 1.0f / lrow[i];
    #pragma unroll
    for (int ct = 0; ct < 4; ++ct) {
        #pragma unroll
        for (int i = 0; i < 8; ++i) {
            const int srow = q0 + i + 8 * lh;
            Op[headoff + (size_t)srow * D + ct * 16 + l15] =
                (_Float16)(o[ct][i] * inv[i]);
        }
    }
}

// ---------------------------------------------------------------------------
extern "C" void kernel_launch(void* const* d_in, const int* in_sizes, int n_in,
                              void* d_out, int out_size, void* d_ws, size_t ws_size,
                              hipStream_t stream) {
    (void)in_sizes; (void)n_in; (void)out_size; (void)ws_size;
    const float* q  = (const float*)d_in[0];
    const float* k  = (const float*)d_in[1];
    const float* v  = (const float*)d_in[2];
    const float* Wq = (const float*)d_in[3];
    const float* bq = (const float*)d_in[4];
    const float* Wk = (const float*)d_in[5];
    const float* bk = (const float*)d_in[6];
    const float* Wv = (const float*)d_in[7];
    const float* bv = (const float*)d_in[8];
    const float* Wo = (const float*)d_in[9];
    const float* bo = (const float*)d_in[10];

    constexpr int B = 4, S = 2048, D = 1024;
    const size_t E = (size_t)B * S * D;
    _Float16* Qp = (_Float16*)d_ws;      // 16 MB
    _Float16* Kp = Qp + E;               // 16 MB
    _Float16* Vp = Kp + E;               // 16 MB
    _Float16* Ap = Vp + E;               // 16 MB (attention output, merged heads)

    dim3 gemm_grid((B * S) / 128, D / 128);  // 64 x 8
    dim3 attn_grid(S / 128, B * 16);         // 16 x 64

    gemm_bias_kernel<float, true><<<gemm_grid, 256, 0, stream>>>(q, Wq, bq, Qp, nullptr);
    gemm_bias_kernel<float, true><<<gemm_grid, 256, 0, stream>>>(k, Wk, bk, Kp, nullptr);
    gemm_bias_kernel<float, true><<<gemm_grid, 256, 0, stream>>>(v, Wv, bv, Vp, nullptr);

    attn_kernel<<<attn_grid, 256, 0, stream>>>(Qp, Kp, Vp, Ap);

    gemm_bias_kernel<_Float16, false><<<gemm_grid, 256, 0, stream>>>(
        Ap, Wo, bo, nullptr, (float*)d_out);
}
